// SparseDepthWaveProgressiveDecoder_50852412785487
// MI455X (gfx1250) — compile-verified, matrix operand order checked by reading, not run on hardware
//
#include <hip/hip_runtime.h>

// ---------------------------------------------------------------------------
// CDNA5 (gfx1250) sparse wavelet depth decoder.
// Convs: implicit-GEMM over v_wmma_f32_16x16x32_bf16 (wave32).
//  * Activations carried in bf16, CHANNEL-LAST ([H][W][C]): each lane's A
//    fragment is two 8-contiguous bf16 runs -> 2x b128 loads, no converts.
//  * Weights repacked tap-major bf16 ([Cout][9][Cin]): each 32-K chunk has a
//    constant conv tap (reflect/index math hoisted); B fragment = 2x b128.
//  * N-tiling NT in {1,2,4}: one A fragment feeds up to 4 WMMAs.
//  * f32 accumulate + f32 bias/activation; stores bf16 (or f32 for the yl
//    head feeding the wavelet chain).
// ---------------------------------------------------------------------------

typedef __attribute__((ext_vector_type(16))) __bf16 v16bf;
typedef __attribute__((ext_vector_type(8)))  __bf16 v8bf;
typedef __attribute__((ext_vector_type(8)))  float  v8f;

#define DEVINL __device__ __forceinline__

DEVINL int refl1(int i, int n) {            // reflect pad=1 (jnp.pad 'reflect')
  if (i < 0) i = -i;
  if (i >= n) i = 2 * n - 2 - i;
  return i;
}

// Repack conv weights: w[Cout][Cin][KK] (f32) -> wp[Cout][KK][Cin] (bf16).
__global__ void repack_kernel(const float* __restrict__ w, __bf16* __restrict__ wp,
                              int Cout, int Cin, int KK) {
  const int total = Cout * Cin * KK;
  int idx = blockIdx.x * blockDim.x + threadIdx.x;
  if (idx >= total) return;
  const int ci = idx % Cin;
  const int t  = idx / Cin;
  const int r  = t % KK;
  const int co = t / KK;
  wp[idx] = (__bf16)w[((size_t)co * Cin + ci) * KK + r];
}

// NCHW f32 (planar) -> NHWC bf16 (channel-last)
__global__ void nchw2nhwc_kernel(const float* __restrict__ src, __bf16* __restrict__ dst,
                                 int C, int HW) {
  const int total = C * HW;
  int idx = blockIdx.x * blockDim.x + threadIdx.x;   // idx = p*C + c (dst order)
  if (idx >= total) return;
  const int c = idx % C;
  const int p = idx / C;
  dst[idx] = (__bf16)src[(size_t)c * HW + p];
}

// act: 0=none  1=ELU  2=leaky(0.1)  3=scale*sigmoid
// x: channel-last bf16 [H][W][Cin]; y: channel-last [H][W][Cout] (OutT).
// Requirements (hold for every conv here): H*W % 16 == 0, Cin % 32 == 0.
template <int KS, int NT, typename OutT>
__global__ void conv_wmma_kernel(const __bf16* __restrict__ x,
                                 const __bf16* __restrict__ wp,
                                 const float* __restrict__ bias,
                                 OutT* __restrict__ y,
                                 int Cin, int H, int W, int Cout,
                                 int act, float scale) {
  const int HW = H * W;
  const int K  = Cin * KS * KS;
  const int tilesM = HW >> 4;
  const int tilesN = (Cout + 16 * NT - 1) / (16 * NT);
  const int wave = (int)((blockIdx.x * blockDim.x + threadIdx.x) >> 5);
  if (wave >= tilesM * tilesN) return;
  const int lane = threadIdx.x & 31;
  const int tM = wave % tilesM;
  const int tG = wave / tilesM;
  const int half = lane >> 4;               // K-half select (A/B), M+8 select (D)
  const int r16  = lane & 15;

  // A rows: this lane's output pixel (M = r16 within tile)
  const int m  = tM * 16 + r16;
  const int ph = m / W;
  const int pw = m - ph * W;

  // B rows (one output channel per lane per n-tile), tap-major bf16 layout
  const __bf16* wrow[NT];
#pragma unroll
  for (int nt = 0; nt < NT; ++nt) {
    const int n  = (tG * NT + nt) * 16 + r16;
    const int nc = (n < Cout) ? n : (Cout - 1);
    wrow[nt] = wp + (size_t)nc * K;
  }

  v8f acc[NT];
#pragma unroll
  for (int nt = 0; nt < NT; ++nt)
#pragma unroll
    for (int i = 0; i < 8; ++i) acc[nt][i] = 0.f;

  const int kofflo = half * 8;              // per-lane K base within chunk

  int tap = 0, cib = 0;                     // k = tap*Cin + ci ; chunk-constant tap
  for (int kb = 0; kb < K; kb += 32) {
    // ---- per-chunk A base (reflect math once per 32 K) ----
    const __bf16* xb;
    if (KS == 3) {
      const int dy = tap / 3;
      const int yy = refl1(ph + dy - 1, H);
      const int xx = refl1(pw + (tap - dy * 3) - 1, W);
      xb = x + (size_t)(yy * W + xx) * Cin + cib + kofflo;
    } else {
      xb = x + (size_t)m * Cin + kb + kofflo;
    }
    // ---- A fragment: two 8-contiguous bf16 runs -> 2x b128, no cvt ----
    const v8bf alo = *reinterpret_cast<const v8bf*>(xb);        // 16B aligned
    const v8bf ahi = *reinterpret_cast<const v8bf*>(xb + 16);
    const v16bf afrag = __builtin_shufflevector(
        alo, ahi, 0, 1, 2, 3, 4, 5, 6, 7, 8, 9, 10, 11, 12, 13, 14, 15);
    // ---- NT WMMAs reuse the A fragment ----
#pragma unroll
    for (int nt = 0; nt < NT; ++nt) {
      const __bf16* p0 = wrow[nt] + kb + kofflo;
      const v8bf lo = *reinterpret_cast<const v8bf*>(p0);       // 16B aligned
      const v8bf hi = *reinterpret_cast<const v8bf*>(p0 + 16);
      const v16bf bfrag = __builtin_shufflevector(
          lo, hi, 0, 1, 2, 3, 4, 5, 6, 7, 8, 9, 10, 11, 12, 13, 14, 15);
      acc[nt] = __builtin_amdgcn_wmma_f32_16x16x32_bf16(false, afrag, false, bfrag,
                                                        (short)0, acc[nt], false, false);
    }
    cib += 32;
    if (KS == 3 && cib == Cin) { cib = 0; ++tap; }
  }

  // ---- D store: channel-last, coalesced across the half-wave ----
#pragma unroll
  for (int nt = 0; nt < NT; ++nt) {
    const int n = (tG * NT + nt) * 16 + r16;
    if (n < Cout) {
      const float bv = bias[n];
#pragma unroll
      for (int v = 0; v < 8; ++v) {
        const int mm = tM * 16 + half * 8 + v;
        float val = acc[nt][v] + bv;
        if (act == 1)      val = (val > 0.f) ? val : (__expf(val) - 1.f);
        else if (act == 2) val = (val > 0.f) ? val : 0.1f * val;
        else if (act == 3) val = scale / (1.f + __expf(-val));
        y[(size_t)mm * Cout + n] = (OutT)val;
      }
    }
  }
}

// out[2Hs x 2Ws x (Cx+Cf)] bf16 = concat(upsample2(x_bf16[Hs x Ws x Cx]),
//                                        f_nchw_f32[Cf x 2Hs x 2Ws])
__global__ void upcat_kernel(const __bf16* __restrict__ x, int Cx, int Hs, int Ws,
                             const float* __restrict__ f, int Cf,
                             __bf16* __restrict__ out) {
  const int H = 2 * Hs, W = 2 * Ws;
  const int Ct = Cx + Cf;
  const int total = H * W * Ct;
  int idx = blockIdx.x * blockDim.x + threadIdx.x;   // idx = p*Ct + c
  if (idx >= total) return;
  const int c = idx % Ct;
  const int p = idx / Ct;
  const int h = p / W;
  const int w = p - h * W;
  __bf16 v;
  if (c < Cx) v = x[(size_t)((h >> 1) * Ws + (w >> 1)) * Cx + c];
  else        v = (__bf16)f[(size_t)(c - Cx) * H * W + p];
  out[idx] = v;
}

// thresh = (max(yl) - min(yl)) * 0.05, single-workgroup tree reduction
__global__ void minmax_thresh_kernel(const float* __restrict__ yl, int n,
                                     float* __restrict__ thresh) {
  __shared__ float smin[1024];
  __shared__ float smax[1024];
  float mn = 3.4e38f, mx = -3.4e38f;
  for (int i = threadIdx.x; i < n; i += blockDim.x) {
    const float v = yl[i];
    mn = fminf(mn, v);
    mx = fmaxf(mx, v);
  }
  smin[threadIdx.x] = mn;
  smax[threadIdx.x] = mx;
  __syncthreads();
  for (int s = blockDim.x >> 1; s > 0; s >>= 1) {
    if ((int)threadIdx.x < s) {
      smin[threadIdx.x] = fminf(smin[threadIdx.x], smin[threadIdx.x + s]);
      smax[threadIdx.x] = fmaxf(smax[threadIdx.x], smax[threadIdx.x + s]);
    }
    __syncthreads();
  }
  if (threadIdx.x == 0) *thresh = (smax[0] - smin[0]) * 0.05f;
}

// yh[3 x H x W] (planar f32) = scale * (d1 - d2) * wmask.
// d1/d2 are channel-last bf16 [H*W][3]; wmask = upsample2(mask(prev planar yh)).
__global__ void yh_kernel(const __bf16* __restrict__ d1, const __bf16* __restrict__ d2,
                          const float* __restrict__ yh_prev,
                          const float* __restrict__ thresh_p,
                          float scale, int H, int W, int use_mask,
                          float* __restrict__ yh) {
  const int HW = H * W;
  const int total = 3 * HW;
  int idx = blockIdx.x * blockDim.x + threadIdx.x;
  if (idx >= total) return;
  const int c = idx / HW;
  const int p = idx - c * HW;
  float maskv = 1.0f;
  if (use_mask) {
    const float thresh = *thresh_p;
    const int h = p / W, w = p - (p / W) * W;
    const int Wp = W >> 1;
    const int HWp = (H >> 1) * Wp;
    const int pp = (h >> 1) * Wp + (w >> 1);
    const float a0 = fabsf(yh_prev[pp]);
    const float a1 = fabsf(yh_prev[HWp + pp]);
    const float a2 = fabsf(yh_prev[2 * HWp + pp]);
    maskv = (fmaxf(fmaxf(a0, a1), a2) > thresh) ? 1.0f : 0.0f;
  }
  yh[idx] = scale * ((float)d1[p * 3 + c] - (float)d2[p * 3 + c]) * maskv;
}

// Haar inverse wavelet transform: (yl[HxW], yh[3xHxW] planar) -> out[2H x 2W]
__global__ void iwt_kernel(const float* __restrict__ yl, const float* __restrict__ yh,
                           int H, int W, float* __restrict__ out) {
  const int HW = H * W;
  int p = blockIdx.x * blockDim.x + threadIdx.x;
  if (p >= HW) return;
  const int h = p / W, w = p - h * W;
  const float l  = yl[p];
  const float h0 = yh[p];
  const float h1 = yh[HW + p];
  const float h2 = yh[2 * HW + p];
  const float c00 = (l + h0 + h1 + h2) * 0.5f;
  const float c01 = (l + h0 - h1 - h2) * 0.5f;
  const float c10 = (l - h0 + h1 - h2) * 0.5f;
  const float c11 = (l - h0 - h1 + h2) * 0.5f;
  const int W2 = 2 * W;
  float* o = out + (size_t)(2 * h) * W2 + 2 * w;
  o[0] = c00;
  o[1] = c01;
  o[W2] = c10;
  o[W2 + 1] = c11;
}

__global__ void clip_kernel(const float* __restrict__ yl, float inv_scale, int n,
                            float* __restrict__ out) {
  int i = blockIdx.x * blockDim.x + threadIdx.x;
  if (i >= n) return;
  out[i] = fminf(fmaxf(yl[i] * inv_scale, 0.0f), 1.0f);
}

// ---------------------------------------------------------------------------
// Host-side orchestration
// ---------------------------------------------------------------------------
static inline int ceil_div(int a, int b) { return (a + b - 1) / b; }

struct RunCtx {
  hipStream_t s;
  const float* const* P;   // flattened d_in pointers
  __bf16* wpk;             // packed-weight arena in d_ws
  size_t cur;              // arena cursor (bf16 elements), deterministic per call
};

// Repack weights (f32 [Cout][Cin][ks*ks] -> bf16 [Cout][ks*ks][Cin]) then conv.
// out_f32: only used for the yl head (ks=3, Cout=1 -> NT=1).
static void conv_op(RunCtx& c, int wi, int bi, const __bf16* x, void* y,
                    int Cin, int H, int W, int Cout, int ks, int act, float scale,
                    bool out_f32 = false) {
  const int KK = ks * ks;
  const size_t sz = (size_t)Cout * Cin * KK;
  __bf16* wp = c.wpk + c.cur;
  c.cur += (sz + 63) & ~(size_t)63;        // keep rows 128B aligned
  repack_kernel<<<ceil_div((int)sz, 256), 256, 0, c.s>>>(c.P[wi], wp, Cout, Cin, KK);

  const int nt = (Cout % 64 == 0) ? 4 : (Cout % 32 == 0) ? 2 : 1;
  const int tiles = (H * W / 16) * ((Cout + 16 * nt - 1) / (16 * nt));
  const int blocks = (tiles + 7) / 8;      // 8 waves / 256-thread block
  const float* b = c.P[bi];
  if (out_f32) {                           // yl head only: ks=3, nt=1
    conv_wmma_kernel<3, 1, float><<<blocks, 256, 0, c.s>>>(
        x, wp, b, (float*)y, Cin, H, W, Cout, act, scale);
    return;
  }
  __bf16* yb = (__bf16*)y;
#define CONV_L(KS, NT) \
  conv_wmma_kernel<KS, NT, __bf16><<<blocks, 256, 0, c.s>>>(x, wp, b, yb, Cin, H, W, Cout, act, scale)
  if (ks == 3) {
    if (nt == 4) CONV_L(3, 4);
    else if (nt == 2) CONV_L(3, 2);
    else CONV_L(3, 1);
  } else {
    if (nt == 4) CONV_L(1, 4);
    else if (nt == 2) CONV_L(1, 2);
    else CONV_L(1, 1);
  }
#undef CONV_L
}

extern "C" void kernel_launch(void* const* d_in, const int* in_sizes, int n_in,
                              void* d_out, int out_size, void* d_ws, size_t ws_size,
                              hipStream_t stream) {
  (void)in_sizes; (void)n_in; (void)ws_size; (void)out_size;

  const float* f0 = (const float*)d_in[0];   // 64 x 192 x 640  (NCHW f32)
  const float* f1 = (const float*)d_in[1];   // 64 x  96 x 320
  const float* f2 = (const float*)d_in[2];   // 128 x 48 x 160
  const float* f3 = (const float*)d_in[3];   // 256 x 24 x  80
  const float* f4 = (const float*)d_in[4];   // 512 x 12 x  40
  static_assert(sizeof(__bf16) == 2, "bf16");
  const float* P[64];
  for (int i = 5; i < 57; ++i) P[i] = (const float*)d_in[i];

  // Param indices (setup_inputs dict insertion order, tuples flattened):
  // i=4: 5/6 upconv_4_0, 7/8 upconv_4_1, 9..12 waveconv_4_0(w1,b1,w2,b2),
  //      13..16 waveconv_4_p1, 17..20 waveconv_4_m1
  // i=3: 21/22, 23/24, 25..28 p1, 29..32 m1
  // i=2: 33/34, 35/36, 37..40 p1, 41..44 m1
  // i=1: 45/46, 47/48, 49..52 p1, 53..56 m1

  // Workspace layout: bf16 activation arena + f32 wavelet chain (~56 MB).
  __bf16* wsb  = (__bf16*)d_ws;
  __bf16* CONV0 = wsb;                     // up to 983040      (bf16, ch-last)
  __bf16* CAT   = CONV0 + 983040;          // up to 11796480    (bf16, ch-last)
  __bf16* UX    = CAT   + 11796480;        // up to 3932160     (bf16, ch-last)
  __bf16* MID   = UX    + 3932160;         // up to 3932160     (bf16, ch-last)
  __bf16* D1    = MID   + 3932160;         // up to 368640      (bf16, [HW][3])
  __bf16* D2    = D1    + 368640;          // up to 368640      (bf16, [HW][3])
  __bf16* F4T   = D2    + 368640;          // 245760            (bf16, ch-last)
  __bf16* WPK   = F4T   + 245760;          // ~3.4M bf16 packed weights
  float*  YH_A  = (float*)(WPK + 3500032); // up to 368640      (f32, planar 3xHW)
  float*  YH_B  = YH_A + 368640;           // up to 368640
  float*  YL_A  = YH_B + 368640;           // up to 491520      (f32, 1 channel)
  float*  YL_B  = YL_A + 491520;           // up to 491520
  float*  SCAL  = YL_B + 491520;           // 16 floats

  RunCtx c{stream, P, WPK, 0};
  const int T = 256;

  // ============================ level i = 4 ==============================
  nchw2nhwc_kernel<<<ceil_div(512 * 480, T), T, 0, stream>>>(f4, F4T, 512, 480);
  conv_op(c, 5, 6, F4T, CONV0, 512, 12, 40, 256, 3, /*ELU*/1, 1.f);
  upcat_kernel<<<ceil_div(512 * 24 * 80, T), T, 0, stream>>>(CONV0, 256, 12, 40, f3, 256, CAT);
  conv_op(c, 7, 8, CAT, UX, 512, 24, 80, 256, 3, 1, 1.f);
  // waveconv_4_0 -> yl = 16 * sigmoid(.)  (f32 output into wavelet chain)
  conv_op(c, 9, 10, UX, MID, 256, 24, 80, 64, 1, /*leaky*/2, 1.f);
  conv_op(c, 11, 12, MID, YL_A, 64, 24, 80, 1, 3, /*sig*/3, 16.f, /*out_f32=*/true);
  // waveconv_4_p1 / m1 -> sigmoid outputs (bf16 [HW][3])
  conv_op(c, 13, 14, UX, MID, 256, 24, 80, 256, 1, 2, 1.f);
  conv_op(c, 15, 16, MID, D1, 256, 24, 80, 3, 3, 3, 1.f);
  conv_op(c, 17, 18, UX, MID, 256, 24, 80, 256, 1, 2, 1.f);
  conv_op(c, 19, 20, MID, D2, 256, 24, 80, 3, 3, 3, 1.f);
  yh_kernel<<<ceil_div(3 * 24 * 80, T), T, 0, stream>>>(D1, D2, YH_A, SCAL, 8.f, 24, 80, 0, YH_A);
  iwt_kernel<<<ceil_div(24 * 80, T), T, 0, stream>>>(YL_A, YH_A, 24, 80, YL_B);   // yl: 48x160

  // ============================ level i = 3 ==============================
  conv_op(c, 21, 22, UX, CONV0, 256, 24, 80, 128, 3, 1, 1.f);
  upcat_kernel<<<ceil_div(256 * 48 * 160, T), T, 0, stream>>>(CONV0, 128, 24, 80, f2, 128, CAT);
  conv_op(c, 23, 24, CAT, UX, 256, 48, 160, 128, 3, 1, 1.f);
  minmax_thresh_kernel<<<1, 1024, 0, stream>>>(YL_B, 48 * 160, SCAL);
  conv_op(c, 25, 26, UX, MID, 128, 48, 160, 128, 1, 2, 1.f);
  conv_op(c, 27, 28, MID, D1, 128, 48, 160, 3, 3, 3, 1.f);
  conv_op(c, 29, 30, UX, MID, 128, 48, 160, 128, 1, 2, 1.f);
  conv_op(c, 31, 32, MID, D2, 128, 48, 160, 3, 3, 3, 1.f);
  yh_kernel<<<ceil_div(3 * 48 * 160, T), T, 0, stream>>>(D1, D2, YH_A, SCAL, 4.f, 48, 160, 1, YH_B);
  iwt_kernel<<<ceil_div(48 * 160, T), T, 0, stream>>>(YL_B, YH_B, 48, 160, YL_A); // yl: 96x320

  // ============================ level i = 2 ==============================
  conv_op(c, 33, 34, UX, CONV0, 128, 48, 160, 64, 3, 1, 1.f);
  upcat_kernel<<<ceil_div(128 * 96 * 320, T), T, 0, stream>>>(CONV0, 64, 48, 160, f1, 64, CAT);
  conv_op(c, 35, 36, CAT, UX, 128, 96, 320, 64, 3, 1, 1.f);
  minmax_thresh_kernel<<<1, 1024, 0, stream>>>(YL_A, 96 * 320, SCAL);
  conv_op(c, 37, 38, UX, MID, 64, 96, 320, 64, 1, 2, 1.f);
  conv_op(c, 39, 40, MID, D1, 64, 96, 320, 3, 3, 3, 1.f);
  conv_op(c, 41, 42, UX, MID, 64, 96, 320, 64, 1, 2, 1.f);
  conv_op(c, 43, 44, MID, D2, 64, 96, 320, 3, 3, 3, 1.f);
  yh_kernel<<<ceil_div(3 * 96 * 320, T), T, 0, stream>>>(D1, D2, YH_B, SCAL, 2.f, 96, 320, 1, YH_A);
  iwt_kernel<<<ceil_div(96 * 320, T), T, 0, stream>>>(YL_A, YH_A, 96, 320, YL_B); // yl: 192x640

  // ============================ level i = 1 ==============================
  conv_op(c, 45, 46, UX, CONV0, 64, 96, 320, 32, 3, 1, 1.f);
  upcat_kernel<<<ceil_div(96 * 192 * 640, T), T, 0, stream>>>(CONV0, 32, 96, 320, f0, 64, CAT);
  conv_op(c, 47, 48, CAT, UX, 96, 192, 640, 32, 3, 1, 1.f);
  minmax_thresh_kernel<<<1, 1024, 0, stream>>>(YL_B, 192 * 640, SCAL);
  conv_op(c, 49, 50, UX, MID, 32, 192, 640, 32, 1, 2, 1.f);
  conv_op(c, 51, 52, MID, D1, 32, 192, 640, 3, 3, 3, 1.f);
  conv_op(c, 53, 54, UX, MID, 32, 192, 640, 32, 1, 2, 1.f);
  conv_op(c, 55, 56, MID, D2, 32, 192, 640, 3, 3, 3, 1.f);
  yh_kernel<<<ceil_div(3 * 192 * 640, T), T, 0, stream>>>(D1, D2, YH_A, SCAL, 1.f, 192, 640, 1, YH_B);
  iwt_kernel<<<ceil_div(192 * 640, T), T, 0, stream>>>(YL_B, YH_B, 192, 640, YL_A); // 384x1280

  // disp = clip(yl / 2^0, 0, 1)
  clip_kernel<<<ceil_div(384 * 1280, T), T, 0, stream>>>(YL_A, 1.0f, 384 * 1280, (float*)d_out);
}
